// DQNDecision_2534030705098
// MI455X (gfx1250) — matrix operand order repaired and checked by reading, not compiled, not run on hardware
//
#include <hip/hip_runtime.h>
#include <hip/hip_bf16.h>
#include <stdint.h>

typedef __attribute__((ext_vector_type(16))) _Float16 v16h;
typedef __attribute__((ext_vector_type(8)))  float    v8f;

#define QN      4096
#define NN      64
#define SS      64
#define DD      320
#define STATE_K 352        // 328 padded to 11 K-tiles of 32
#define HIDN    128
#define QT      16         // queries per workgroup (WMMA M dim)
#define NEGV    (-1e9f)

// workspace layout (in _Float16 elements): WMMA-B fragment-native tiles
#define W1_KT  11
#define W1_NT  8
#define W1_OFF  0
#define W2_OFF  (W1_KT*W1_NT*512)            // 45056
#define WH1_OFF (W2_OFF + 4*8*512)           // 61440
#define WH2_OFF (WH1_OFF + 4*8*512)          // 77824

// register-file fragment table: 22 (L1) + 8 (L2) + 8 (L3) + 4 (L4)
#define FRAG_L1 0
#define FRAG_L2 22
#define FRAG_L3 30
#define FRAG_L4 38
#define NFRAG   42

// ---------------------------------------------------------------------------
// Pack fp32 row-major W[K][N] into f16 B-fragment layout:
// dst[tile=kt*ntiles+nt][lane][h] with n = nt*16 + lane%16,
//                                     k = kt*32 + (lane>=16)*16 + h
// ---------------------------------------------------------------------------
__global__ void pack_wmma_b(const float* __restrict__ W, _Float16* __restrict__ dst,
                            int Kact, int ktiles, int ntiles) {
    int idx = blockIdx.x * blockDim.x + threadIdx.x;
    int total = ktiles * ntiles * 512;
    if (idx >= total) return;
    int tile   = idx >> 9;
    int within = idx & 511;
    int lane   = within >> 4;
    int h      = within & 15;
    int kt = tile / ntiles;
    int nt = tile % ntiles;
    int n = nt * 16 + (lane & 15);
    int k = kt * 32 + ((lane >> 4) << 4) + h;
    float v = (k < Kact) ? W[(size_t)k * (ntiles * 16) + n] : 0.0f;
    dst[idx] = (_Float16)v;
}

__device__ __forceinline__ float silu_f(float x) { return x / (1.0f + __expf(-x)); }

// gfx1250 async global->LDS copy, 16B per lane, tracked by ASYNCcnt.
// No memory clobber on issue: destination buffer is only read after the
// (volatile, memory-clobbering) wait + barrier, so ordering is preserved
// without blocking register residency of unrelated values.
__device__ __forceinline__ void async_g2l_b128(uint32_t lds_off, const void* gaddr) {
    asm volatile("global_load_async_to_lds_b128 %0, %1, off"
                 :: "v"(lds_off), "v"(gaddr));
}
__device__ __forceinline__ void async_wait0() {
    asm volatile("s_wait_asynccnt 0x0" ::: "memory");
}

__device__ __forceinline__ v16h loadBfrag(const _Float16* __restrict__ Bws, int tile, int lane) {
    return *(const v16h*)(Bws + (size_t)tile * 512 + lane * 16);
}

// One layer of 16xK @ KxN; B-fragments come from the register table wB with
// compile-time indices (BASE + kt*(NT/4) + j) so they stay in VGPRs.
template <int KT, int NT, int BASE>
__device__ __forceinline__ void gemm_reg(const _Float16* __restrict__ A, int apitch,
                                         int lane, const v16h (&wB)[NFRAG],
                                         v8f (&acc)[NT / 4]) {
#pragma unroll
    for (int j = 0; j < NT / 4; ++j) acc[j] = (v8f){};
    const int m   = lane & 15;
    const int grp = (lane >> 4) << 4;     // A: lanes 16-31 start at K+16
#pragma unroll
    for (int kt = 0; kt < KT; ++kt) {
        const _Float16* p = A + m * apitch + kt * 32 + grp;
        v16h a;
#pragma unroll
        for (int h = 0; h < 8; ++h) { a[h] = p[h]; a[h + 8] = p[h + 8]; }
#pragma unroll
        for (int j = 0; j < NT / 4; ++j)
            acc[j] = __builtin_amdgcn_wmma_f32_16x16x32_f16(
                false, a, false, wB[BASE + kt * (NT / 4) + j], (short)0,
                acc[j], false, false);
    }
}

// ---------------------------------------------------------------------------
// Main kernel: one block = 16 queries, all 64 scan steps.
// Weights register-resident; double-buffered async gather overlapped with WMMA.
// ---------------------------------------------------------------------------
__global__ void __launch_bounds__(128)
dqn_main(const float* __restrict__ tasks, const float* __restrict__ cons,
         const int* __restrict__ masks, const int* __restrict__ topos,
         const float* __restrict__ b1, const float* __restrict__ b2,
         const float* __restrict__ bh1, const float* __restrict__ bh2,
         const _Float16* __restrict__ wsB, short* __restrict__ out) {
    __shared__ _Float16 sState[QT][STATE_K];   // 11264 B (A of layer 1)
    __shared__ _Float16 sAct[2][QT][HIDN];     //  8192 B (ping-pong activations)
    __shared__ float    sTask[2][QT][DD];      // 40960 B (double-buffered raw rows)
    __shared__ int      sMask[2][QT][SS];      //  8192 B (double-buffered masks)
    __shared__ int      sTopoAll[QT][NN];      //  4096 B (reversed topo order)
    __shared__ float    sQos0[QT][NN];         //  4096 B (qos channel 0)
    __shared__ float    sQv[QT][SS];           //  4096 B
    __shared__ float    sRet[QT][NN];          //  4096 B
    __shared__ float    sB1[HIDN], sB2[HIDN], sBh1[HIDN], sBh2[SS];
    __shared__ float    sAvail[QT], sThr[QT], sRel[QT], sRt[QT];

    const int t     = threadIdx.x;
    const int lane  = t & 31;
    const int wave  = t >> 5;
    const int qbase = blockIdx.x * QT;

    // ---- load all owned weight B-fragments into registers (once) ----
    v16h wB[NFRAG];
#pragma unroll
    for (int kt = 0; kt < W1_KT; ++kt)
#pragma unroll
        for (int j = 0; j < 2; ++j)
            wB[FRAG_L1 + kt * 2 + j] = loadBfrag(wsB + W1_OFF, kt * 8 + wave + 4 * j, lane);
#pragma unroll
    for (int kt = 0; kt < 4; ++kt)
#pragma unroll
        for (int j = 0; j < 2; ++j) {
            wB[FRAG_L2 + kt * 2 + j] = loadBfrag(wsB + W2_OFF,  kt * 8 + wave + 4 * j, lane);
            wB[FRAG_L3 + kt * 2 + j] = loadBfrag(wsB + WH1_OFF, kt * 8 + wave + 4 * j, lane);
        }
#pragma unroll
    for (int kt = 0; kt < 4; ++kt)
        wB[FRAG_L4 + kt] = loadBfrag(wsB + WH2_OFF, kt * 4 + wave, lane);

    // ---- one-time init ----
    for (int i = t; i < HIDN; i += 128) { sB1[i] = b1[i]; sB2[i] = b2[i]; sBh1[i] = bh1[i]; }
    for (int i = t; i < SS; i += 128) sBh2[i] = bh2[i];
    for (int i = t; i < QT * NN; i += 128) {
        (&sQos0[0][0])[i] = -3.0f;
        (&sRet[0][0])[i]  = 0.0f;
        int qi = i >> 6, s = i & 63;                  // reversed topological order
        sTopoAll[qi][s] = topos[(size_t)(qbase + qi) * NN + (NN - 1 - s)];
    }
    if (t < QT) { sAvail[t] = 1.0f; sThr[t] = 3.0f; sRel[t] = 1.0f; }
    for (int i = t; i < QT * 4; i += 128) {            // constraints: cols 320..323, constant
        int qi = i >> 2, c = i & 3;
        sState[qi][DD + c] = (_Float16)cons[(size_t)(qbase + qi) * 4 + c];
    }
    for (int i = t; i < QT * (STATE_K - 328); i += 128) {  // zero pad cols 328..351
        int qi = i / (STATE_K - 328), c = i % (STATE_K - 328);
        sState[qi][328 + c] = (_Float16)0.0f;
    }
    __syncthreads();

    // ---- prologue: async gather of step 0 rows into buffer 0 ----
    {
        for (int i = t; i < QT * (DD / 4); i += 128) {        // 80 x 16B chunks / row
            int qi = i / (DD / 4), c = i % (DD / 4);
            int node = sTopoAll[qi][0];
            const float* g = tasks + ((size_t)(qbase + qi) * NN + node) * DD + c * 4;
            async_g2l_b128((uint32_t)(uintptr_t)&sTask[0][qi][c * 4], g);
        }
        for (int i = t; i < QT * (SS / 4); i += 128) {        // 16 x 16B chunks / row
            int qi = i / (SS / 4), c = i % (SS / 4);
            int node = sTopoAll[qi][0];
            const int* g = masks + ((size_t)(qbase + qi) * NN + node) * SS + c * 4;
            async_g2l_b128((uint32_t)(uintptr_t)&sMask[0][qi][c * 4], g);
        }
        async_wait0();
        __syncthreads();
    }

#pragma unroll 1
    for (int step = 0; step < NN; ++step) {
        const int buf = step & 1;

        // ---- kick off async gather for step+1 into the other buffer ----
        if (step + 1 < NN) {
            const int nb = buf ^ 1;
            for (int i = t; i < QT * (DD / 4); i += 128) {
                int qi = i / (DD / 4), c = i % (DD / 4);
                int node = sTopoAll[qi][step + 1];
                const float* g = tasks + ((size_t)(qbase + qi) * NN + node) * DD + c * 4;
                async_g2l_b128((uint32_t)(uintptr_t)&sTask[nb][qi][c * 4], g);
            }
            for (int i = t; i < QT * (SS / 4); i += 128) {
                int qi = i / (SS / 4), c = i % (SS / 4);
                int node = sTopoAll[qi][step + 1];
                const int* g = masks + ((size_t)(qbase + qi) * NN + node) * SS + c * 4;
                async_g2l_b128((uint32_t)(uintptr_t)&sMask[nb][qi][c * 4], g);
            }
        }

        // ---- convert task row f32->f16 into state tile; recurrence feature ----
        for (int i = t; i < QT * DD; i += 128) {
            int qi = i / DD, d = i % DD;
            sState[qi][d] = (_Float16)sTask[buf][qi][d];
        }
        if (t < QT) {
            float mx = -3.4e38f;
            for (int n = 0; n < NN; ++n) mx = fmaxf(mx, sTask[buf][t][n] * sQos0[t][n]);
            if (step == 0) mx -= 3.0f;
            sRt[t] = mx;
            sState[t][324] = (_Float16)mx;
            sState[t][325] = (_Float16)sAvail[t];
            sState[t][326] = (_Float16)sThr[t];
            sState[t][327] = (_Float16)sRel[t];
        }
        __syncthreads();

        // ---- layer 1: h1 = silu(state @ W1 + b1)  [16x352 @ 352x128] ----
        {
            v8f acc[2];
            gemm_reg<W1_KT, 8, FRAG_L1>(&sState[0][0], STATE_K, lane, wB, acc);
#pragma unroll
            for (int j = 0; j < 2; ++j) {
                int n = (wave + 4 * j) * 16 + (lane & 15);
                float bn = sB1[n];
#pragma unroll
                for (int r = 0; r < 8; ++r)
                    sAct[0][r + ((lane >> 4) << 3)][n] = (_Float16)silu_f(acc[j][r] + bn);
            }
        }
        __syncthreads();
        // ---- layer 2: q_in = silu(h1 @ W2 + b2) ----
        {
            v8f acc[2];
            gemm_reg<4, 8, FRAG_L2>(&sAct[0][0][0], HIDN, lane, wB, acc);
#pragma unroll
            for (int j = 0; j < 2; ++j) {
                int n = (wave + 4 * j) * 16 + (lane & 15);
                float bn = sB2[n];
#pragma unroll
                for (int r = 0; r < 8; ++r)
                    sAct[1][r + ((lane >> 4) << 3)][n] = (_Float16)silu_f(acc[j][r] + bn);
            }
        }
        __syncthreads();
        // ---- layer 3: h2 = silu(q_in @ Wh1 + bh1) ----
        {
            v8f acc[2];
            gemm_reg<4, 8, FRAG_L3>(&sAct[1][0][0], HIDN, lane, wB, acc);
#pragma unroll
            for (int j = 0; j < 2; ++j) {
                int n = (wave + 4 * j) * 16 + (lane & 15);
                float bn = sBh1[n];
#pragma unroll
                for (int r = 0; r < 8; ++r)
                    sAct[0][r + ((lane >> 4) << 3)][n] = (_Float16)silu_f(acc[j][r] + bn);
            }
        }
        __syncthreads();
        // ---- layer 4: qv = h2 @ Wh2 + bh2  [16x128 @ 128x64] ----
        {
            v8f acc[1];
            gemm_reg<4, 4, FRAG_L4>(&sAct[0][0][0], HIDN, lane, wB, acc);
            int n = wave * 16 + (lane & 15);
            float bn = sBh2[n];
#pragma unroll
            for (int r = 0; r < 8; ++r)
                sQv[r + ((lane >> 4) << 3)][n] = acc[0][r] + bn;
        }
        __syncthreads();

        // ---- masked argmax, ret accumulate, recurrence update ----
        if (t < QT) {
            float best = -3.4e38f;
            int ser = 0;
            for (int s = 0; s < SS; ++s) {
                float v = (sMask[buf][t][s] == 0) ? NEGV : sQv[t][s];
                if (v > best) { best = v; ser = s; }
            }
            int topo = sTopoAll[t][step];
            sRet[t][topo] += (float)ser;
            float sq0 = sTask[buf][t][NN + 4 * ser + 0];
            float sq1 = sTask[buf][t][NN + 4 * ser + 1];
            float sq2 = sTask[buf][t][NN + 4 * ser + 2];
            float sq3 = sTask[buf][t][NN + 4 * ser + 3];
            sQos0[t][topo] = sq0 + sRt[t];
            sAvail[t] *= sq1;
            sThr[t] = fminf(sq2, sThr[t]);
            sRel[t] *= sq3;
        }
        // drain this step's async copies (long since covered by the GEMMs)
        async_wait0();
        __syncthreads();
    }

    // ---- write ret as int16 ----
    for (int i = t; i < QT * NN; i += 128) {
        int qi = i >> 6, n = i & 63;
        out[(size_t)(qbase + qi) * NN + n] = (short)sRet[qi][n];
    }
}

// ---------------------------------------------------------------------------
extern "C" void kernel_launch(void* const* d_in, const int* in_sizes, int n_in,
                              void* d_out, int out_size, void* d_ws, size_t ws_size,
                              hipStream_t stream) {
    const float* tasks = (const float*)d_in[0];
    const float* cons  = (const float*)d_in[1];
    const int*   masks = (const int*)d_in[2];
    const int*   topos = (const int*)d_in[3];
    const float* W1  = (const float*)d_in[4];
    const float* b1  = (const float*)d_in[5];
    const float* W2  = (const float*)d_in[6];
    const float* b2  = (const float*)d_in[7];
    const float* Wh1 = (const float*)d_in[8];
    const float* bh1 = (const float*)d_in[9];
    const float* Wh2 = (const float*)d_in[10];
    const float* bh2 = (const float*)d_in[11];
    _Float16* wsB = (_Float16*)d_ws;

    // Pack weights into WMMA-B fragment-native f16 layout (L2-resident afterwards).
    pack_wmma_b<<<(W1_KT * W1_NT * 512) / 256, 256, 0, stream>>>(W1, wsB + W1_OFF, 328, W1_KT, 8);
    pack_wmma_b<<<(4 * 8 * 512) / 256, 256, 0, stream>>>(W2,  wsB + W2_OFF,  128, 4, 8);
    pack_wmma_b<<<(4 * 8 * 512) / 256, 256, 0, stream>>>(Wh1, wsB + WH1_OFF, 128, 4, 8);
    pack_wmma_b<<<(4 * 4 * 512) / 256, 256, 0, stream>>>(Wh2, wsB + WH2_OFF, 128, 4, 4);

    dqn_main<<<QN / QT, 128, 0, stream>>>(tasks, cons, masks, topos,
                                          b1, b2, bh1, bh2, wsB, (short*)d_out);
}
